// time_lstm_6098853561017
// MI455X (gfx1250) — compile-verified, matrix operand order checked by reading, not run on hardware
//
#include <hip/hip_runtime.h>
#include <hip/hip_bf16.h>
#include <cstdint>
#include <cstddef>

// ---------------------------------------------------------------------------
// time-LSTM (dual branch) for MI455X / gfx1250, wave32, bf16 WMMA.
//   T=64, B=512, D=512, U=512, P0=256, P1=128, NC=1
// All GEMM operands are bf16 in memory (weights transposed once per launch,
// activations converted once per production site), so the WMMA inner loop is
// pure global_load_b128 + v_wmma_f32_16x16x32_bf16 with no VALU conversion.
// ---------------------------------------------------------------------------

#define LRELU_ALPHA 0.3f

typedef __attribute__((ext_vector_type(16))) __bf16 v16bf;
typedef __attribute__((ext_vector_type(8)))  float  v8f;

__device__ __forceinline__ unsigned short f2bf(float f) {
  unsigned int u = __float_as_uint(f);
  u += 0x7FFFu + ((u >> 16) & 1u);   // round-to-nearest-even
  return (unsigned short)(u >> 16);
}
__device__ __forceinline__ float bf2f(unsigned short s) {
  return __uint_as_float(((unsigned int)s) << 16);
}
__device__ __forceinline__ float sigf(float x) { return 1.0f / (1.0f + __expf(-x)); }

union Frag {
  v16bf v;
  unsigned short s[16];
  uint4 q[2];
};

// ------------------------------- GEMM ---------------------------------------
// C = act( A[M,K](bf16,row,lda) * Bt[N,K](bf16,row) + bias )
// M is always 512 (grid.y = 4, block tile 128x64, 8 waves, wave tile 32x32)

struct GemmJob {
  const unsigned short* A;    // bf16 activations [M,K] row-major (lda)
  const unsigned short* Bt;   // bf16 transposed weight [N,K]
  float* C;                   // optional f32 output
  unsigned short* Cb;         // optional bf16 output
  const float* bias;          // length N or nullptr
  int lda, N, K, act;         // act: 0 = none, 1 = leaky relu
};
struct GemmPack { GemmJob j[26]; };

__global__ __launch_bounds__(256) void gemm_wmma_bf16(GemmPack p) {
  const GemmJob job = p.j[blockIdx.z];
  const int nbase = blockIdx.x * 64;
  if (nbase >= job.N) return;
  const int mbase = blockIdx.y * 128;
  const int wave = threadIdx.x >> 5;
  const int lane = threadIdx.x & 31;
  const int wm = wave & 3;          // 4 waves along M
  const int wn = wave >> 2;         // 2 waves along N
  const int mRow = mbase + wm * 32;
  const int nCol = nbase + wn * 32;
  const int lrow = lane & 15;
  const int lk   = (lane >> 4) << 3;   // 0 or 8 : K sub-offset per lane half

  v8f acc[2][2] = {};

  for (int kk = 0; kk < job.K; kk += 32) {
    Frag a[2], b[2];
#pragma unroll
    for (int ms = 0; ms < 2; ++ms) {
      const unsigned short* ap =
          job.A + (size_t)(mRow + ms * 16 + lrow) * job.lda + kk + lk;
      a[ms].q[0] = *(const uint4*)(ap);
      a[ms].q[1] = *(const uint4*)(ap + 16);
    }
#pragma unroll
    for (int ns = 0; ns < 2; ++ns) {
      const unsigned short* bp =
          job.Bt + (size_t)(nCol + ns * 16 + lrow) * job.K + kk + lk;
      b[ns].q[0] = *(const uint4*)(bp);
      b[ns].q[1] = *(const uint4*)(bp + 16);
    }
#pragma unroll
    for (int ms = 0; ms < 2; ++ms)
#pragma unroll
      for (int ns = 0; ns < 2; ++ns)
        acc[ms][ns] = __builtin_amdgcn_wmma_f32_16x16x32_bf16(
            false, a[ms].v, false, b[ns].v, (short)0, acc[ms][ns], false, false);
  }

  const int rofs = (lane >> 4) << 3;   // lanes 16..31 hold M rows 8..15
  const int cofs = lane & 15;
#pragma unroll
  for (int ms = 0; ms < 2; ++ms)
#pragma unroll
    for (int ns = 0; ns < 2; ++ns) {
#pragma unroll
      for (int r = 0; r < 8; ++r) {
        int row = mRow + ms * 16 + rofs + r;
        int col = nCol + ns * 16 + cofs;
        float v = acc[ms][ns][r];
        if (job.bias) v += job.bias[col];
        if (job.act) v = v > 0.0f ? v : LRELU_ALPHA * v;
        size_t o = (size_t)row * job.N + col;
        if (job.C)  job.C[o]  = v;
        if (job.Cb) job.Cb[o] = f2bf(v);
      }
    }
}

// --------------------------- weight convert ---------------------------------
// W[K,N] f32 row-major  ->  O[N,K] bf16 bits (transposed)
struct CvtJob { const float* W; unsigned short* O; int K, N; };
struct CvtPack { CvtJob j[32]; };

__global__ __launch_bounds__(256) void wconvert(CvtPack p) {
  CvtJob job = p.j[blockIdx.z];
  int idx = blockIdx.x * 256 + threadIdx.x;
  int total = job.K * job.N;
  if (idx >= total) return;
  int k = idx / job.N;
  int n = idx - k * job.N;
  job.O[(size_t)n * job.K + k] = f2bf(job.W[idx]);
}

// straight f32 -> bf16 copy (one timestep of x: 512*1024 elements, 4/thread)
__global__ __launch_bounds__(256) void xconvert(const float* __restrict__ in,
                                                unsigned short* __restrict__ out) {
  int idx = (blockIdx.x * 256 + threadIdx.x) * 4;
  float4 f = *(const float4*)(in + idx);
  ushort4 o;
  o.x = f2bf(f.x); o.y = f2bf(f.y); o.z = f2bf(f.z); o.w = f2bf(f.w);
  *(ushort4*)(out + idx) = o;
}

// ------------------------------ elementwise ---------------------------------
__global__ __launch_bounds__(256) void zero_state(float* ccf, float* cvf,
                                                  unsigned short* ccb, unsigned short* cvb,
                                                  unsigned short* hcb, unsigned short* hvb) {
  int i = blockIdx.x * 256 + threadIdx.x;
  ccf[i] = 0.0f; cvf[i] = 0.0f;
  ccb[i] = 0;    cvb[i] = 0;
  hcb[i] = 0;    hvb[i] = 0;
}

struct GatePtrs {
  const float *Xi, *Hi, *Ci, *Xf, *Hf, *Cf, *Xt, *Tt, *Xz, *Hz;
};
struct EwGatesArgs {
  GatePtrs c, v;
  float *cc, *cv;                 // f32 cell state, updated in place
  unsigned short *ccb, *cvb;      // bf16 mirror of new cell state
};

__global__ __launch_bounds__(256) void ew_gates(EwGatesArgs a) {
  int i = blockIdx.x * 256 + threadIdx.x;   // exactly B*U = 262144 threads
  {
    float ig = sigf(a.c.Xi[i] + a.c.Hi[i] + a.c.Ci[i]);
    float fg = sigf(a.c.Xf[i] + a.c.Hf[i] + a.c.Cf[i]);
    float tg = sigf(a.c.Xt[i] + sigf(a.c.Tt[i]));
    float zg = tanhf(a.c.Xz[i] + a.c.Hz[i]);
    float nc = fg * a.cc[i] + ig * tg * zg;
    a.cc[i] = nc;
    a.ccb[i] = f2bf(nc);
  }
  {
    float ig = sigf(a.v.Xi[i] + a.v.Hi[i] + a.v.Ci[i]);
    float fg = sigf(a.v.Xf[i] + a.v.Hf[i] + a.v.Cf[i]);
    float tg = sigf(a.v.Xt[i] + sigf(a.v.Tt[i]));
    float zg = tanhf(a.v.Xz[i] + a.v.Hz[i]);
    float nc = fg * a.cv[i] + ig * tg * zg;
    a.cv[i] = nc;
    a.cvb[i] = f2bf(nc);
  }
}

struct EwOutArgs {
  const float *Xoc, *Toc, *Hoc, *Coc;
  const float *Xov, *Tov, *Hov, *Cov;
  const float *cc, *cv;
  unsigned short *hcb, *hvb;      // h only consumed by GEMMs -> bf16 only
};

__global__ __launch_bounds__(256) void ew_out(EwOutArgs a) {
  int i = blockIdx.x * 256 + threadIdx.x;
  float oc = sigf(a.Xoc[i] + a.Toc[i] + a.Hoc[i] + a.Coc[i]);
  a.hcb[i] = f2bf(oc * tanhf(a.cc[i]));
  float ov = sigf(a.Xov[i] + a.Tov[i] + a.Hov[i] + a.Cov[i]);
  a.hvb[i] = f2bf(ov * tanhf(a.cv[i]));
}

// final head: N=1 dot products + sigmoid, fused output write (bf16 inputs)
__global__ __launch_bounds__(256) void head3(const unsigned short* a2c,
                                             const unsigned short* a2v,
                                             const float* wc, const float* bc,
                                             const float* wv, const float* bv,
                                             float* outc, float* outcv) {
  int b = blockIdx.x * 256 + threadIdx.x;
  if (b >= 512) return;
  float sc = bc[0], sv = bv[0];
  const unsigned short* rc = a2c + (size_t)b * 128;
  const unsigned short* rv = a2v + (size_t)b * 128;
#pragma unroll 8
  for (int k = 0; k < 128; ++k) {
    sc += bf2f(rc[k]) * wc[k];
    sv += bf2f(rv[k]) * wv[k];
  }
  float pc = sigf(sc), pv = sigf(sv);
  outc[b] = pc;
  outcv[b] = pc * pv;
}

// ------------------------------- launch -------------------------------------
extern "C" void kernel_launch(void* const* d_in, const int* in_sizes, int n_in,
                              void* d_out, int out_size, void* d_ws, size_t ws_size,
                              hipStream_t stream) {
  (void)in_sizes; (void)n_in; (void)out_size;
  // Input order (dict order): 0:x; 1..20: X_NAMES (W,b)x10 in order
  // xfc,xic,xoc,xzc,xtc,xfv,xiv,xov,xzv,xtv; 21..28: hfc,hic,hoc,hzc,hfv,hiv,hov,hzv;
  // 29..34: cic,cfc,coc,civ,cfv,cov; 35..38: ttc,toc,ttv,tov;
  // 39..44: pc0_W,pc0_b,pc1_W,pc1_b,fc_c_W,fc_c_b; 45..50: pv0_W,...,fc_v_b
  const float* x = (const float*)d_in[0];
  const size_t MM = 512u * 512u;

  if (ws_size < 51000000u) return;   // ~50.8 MB used
  char* wsb = (char*)d_ws;

  // bf16 weight area
  unsigned short* wt0 = (unsigned short*)wsb;
  auto WT = [&](int s) { return wt0 + (size_t)s * MM; };
  unsigned short* pc0t = wt0 + (size_t)28 * MM;
  unsigned short* pv0t = pc0t + 512 * 256;
  unsigned short* pc1t = pv0t + 512 * 256;
  unsigned short* pv1t = pc1t + 256 * 128;
  size_t off = ((size_t)28 * MM + 2u * 512 * 256 + 2u * 256 * 128) * 2u; // 15,335,424

  // f32 gate buffers (28 x 1MB)
  float* gbase = (float*)(wsb + off);
  auto G = [&](int s) { return gbase + (size_t)s * MM; };
  off += (size_t)28 * MM * 4u;

  // f32 cell state
  float* ccf = (float*)(wsb + off); off += MM * 4u;
  float* cvf = (float*)(wsb + off); off += MM * 4u;
  // bf16 activations
  unsigned short* ccb = (unsigned short*)(wsb + off); off += MM * 2u;
  unsigned short* cvb = (unsigned short*)(wsb + off); off += MM * 2u;
  unsigned short* hcb = (unsigned short*)(wsb + off); off += MM * 2u;
  unsigned short* hvb = (unsigned short*)(wsb + off); off += MM * 2u;
  unsigned short* xstep = (unsigned short*)(wsb + off); off += (size_t)512 * 1024 * 2u;
  unsigned short* a1cb = (unsigned short*)(wsb + off); off += (size_t)512 * 256 * 2u;
  unsigned short* a1vb = (unsigned short*)(wsb + off); off += (size_t)512 * 256 * 2u;
  unsigned short* a2cb = (unsigned short*)(wsb + off); off += (size_t)512 * 128 * 2u;
  unsigned short* a2vb = (unsigned short*)(wsb + off); off += (size_t)512 * 128 * 2u;

  // weight slot -> input index
  int wIdx[28];
  for (int s = 0; s < 10; ++s) wIdx[s] = 1 + 2 * s;        // X_NAMES W
  for (int s = 0; s < 8;  ++s) wIdx[10 + s] = 21 + s;      // H_NAMES
  for (int s = 0; s < 6;  ++s) wIdx[18 + s] = 29 + s;      // C_NAMES
  for (int s = 0; s < 4;  ++s) wIdx[24 + s] = 35 + s;      // T_NAMES

  // convert + transpose all weights to bf16
  CvtPack cp{};
  for (int s = 0; s < 28; ++s) cp.j[s] = { (const float*)d_in[wIdx[s]], WT(s), 512, 512 };
  cp.j[28] = { (const float*)d_in[39], pc0t, 512, 256 };
  cp.j[29] = { (const float*)d_in[45], pv0t, 512, 256 };
  cp.j[30] = { (const float*)d_in[41], pc1t, 256, 128 };
  cp.j[31] = { (const float*)d_in[47], pv1t, 256, 128 };
  wconvert<<<dim3(1024, 1, 32), 256, 0, stream>>>(cp);

  zero_state<<<1024, 256, 0, stream>>>(ccf, cvf, ccb, cvb, hcb, hvb);

  // gate buffer slots:
  // 0 Xic 1 Hic 2 Cic 3 Xfc 4 Hfc 5 Cfc 6 Xtc 7 Ttc 8 Xzc 9 Hzc
  // 10 Xoc 11 Toc 12 Hoc 13 Coc; +14 for v branch
  for (int t = 0; t < 64; ++t) {
    // stage this step's x in bf16 (inp = cols 0..511, ts = cols 512..1023)
    xconvert<<<512, 256, 0, stream>>>(x + (size_t)t * 512 * 1024, xstep);
    const unsigned short* xin = xstep;
    const unsigned short* tsn = xstep + 512;

    GemmPack pa{};
    int nj = 0;
    auto add = [&](const unsigned short* A, int lda, int wslot, float* C,
                   const float* bias, int N, int K, int act) {
      pa.j[nj++] = GemmJob{ A, WT(wslot), C, nullptr, bias, lda, N, K, act };
    };
    // x projections (with bias)
    add(xin, 1024, 1, G(0),  (const float*)d_in[4],  512, 512, 0);  // xic
    add(xin, 1024, 0, G(3),  (const float*)d_in[2],  512, 512, 0);  // xfc
    add(xin, 1024, 4, G(6),  (const float*)d_in[10], 512, 512, 0);  // xtc
    add(xin, 1024, 3, G(8),  (const float*)d_in[8],  512, 512, 0);  // xzc
    add(xin, 1024, 2, G(10), (const float*)d_in[6],  512, 512, 0);  // xoc
    add(xin, 1024, 6, G(14), (const float*)d_in[14], 512, 512, 0);  // xiv
    add(xin, 1024, 5, G(17), (const float*)d_in[12], 512, 512, 0);  // xfv
    add(xin, 1024, 9, G(20), (const float*)d_in[20], 512, 512, 0);  // xtv
    add(xin, 1024, 8, G(22), (const float*)d_in[18], 512, 512, 0);  // xzv
    add(xin, 1024, 7, G(24), (const float*)d_in[16], 512, 512, 0);  // xov
    // time projections (no bias)
    add(tsn, 1024, 24, G(7),  nullptr, 512, 512, 0);  // ttc
    add(tsn, 1024, 25, G(11), nullptr, 512, 512, 0);  // toc
    add(tsn, 1024, 26, G(21), nullptr, 512, 512, 0);  // ttv
    add(tsn, 1024, 27, G(25), nullptr, 512, 512, 0);  // tov
    // h projections
    add(hcb, 512, 11, G(1),  nullptr, 512, 512, 0);  // hic
    add(hcb, 512, 10, G(4),  nullptr, 512, 512, 0);  // hfc
    add(hcb, 512, 13, G(9),  nullptr, 512, 512, 0);  // hzc
    add(hcb, 512, 12, G(12), nullptr, 512, 512, 0);  // hoc
    add(hvb, 512, 15, G(15), nullptr, 512, 512, 0);  // hiv
    add(hvb, 512, 14, G(18), nullptr, 512, 512, 0);  // hfv
    add(hvb, 512, 17, G(23), nullptr, 512, 512, 0);  // hzv
    add(hvb, 512, 16, G(26), nullptr, 512, 512, 0);  // hov
    // c projections on OLD cell state
    add(ccb, 512, 18, G(2),  nullptr, 512, 512, 0);  // cic
    add(ccb, 512, 19, G(5),  nullptr, 512, 512, 0);  // cfc
    add(cvb, 512, 21, G(16), nullptr, 512, 512, 0);  // civ
    add(cvb, 512, 22, G(19), nullptr, 512, 512, 0);  // cfv
    gemm_wmma_bf16<<<dim3(8, 4, nj), 256, 0, stream>>>(pa);

    EwGatesArgs eg = {
      { G(0), G(1), G(2), G(3), G(4), G(5), G(6), G(7), G(8), G(9) },
      { G(14), G(15), G(16), G(17), G(18), G(19), G(20), G(21), G(22), G(23) },
      ccf, cvf, ccb, cvb
    };
    ew_gates<<<1024, 256, 0, stream>>>(eg);

    // coc/cov on NEW cell state
    GemmPack pb{};
    pb.j[0] = GemmJob{ ccb, WT(20), G(13), nullptr, nullptr, 512, 512, 512, 0 };
    pb.j[1] = GemmJob{ cvb, WT(23), G(27), nullptr, nullptr, 512, 512, 512, 0 };
    gemm_wmma_bf16<<<dim3(8, 4, 2), 256, 0, stream>>>(pb);

    EwOutArgs eo = { G(10), G(11), G(12), G(13),
                     G(24), G(25), G(26), G(27),
                     ccf, cvf, hcb, hvb };
    ew_out<<<1024, 256, 0, stream>>>(eo);

    // prediction heads (bf16 intermediates, LeakyReLU fused in epilogue)
    GemmPack ph1{};
    ph1.j[0] = GemmJob{ hcb, pc0t, nullptr, a1cb, (const float*)d_in[40], 512, 256, 512, 1 };
    ph1.j[1] = GemmJob{ hvb, pv0t, nullptr, a1vb, (const float*)d_in[46], 512, 256, 512, 1 };
    gemm_wmma_bf16<<<dim3(4, 4, 2), 256, 0, stream>>>(ph1);

    GemmPack ph2{};
    ph2.j[0] = GemmJob{ a1cb, pc1t, nullptr, a2cb, (const float*)d_in[42], 256, 128, 256, 1 };
    ph2.j[1] = GemmJob{ a1vb, pv1t, nullptr, a2vb, (const float*)d_in[48], 256, 128, 256, 1 };
    gemm_wmma_bf16<<<dim3(2, 4, 2), 256, 0, stream>>>(ph2);

    head3<<<2, 256, 0, stream>>>(a2cb, a2vb,
                                 (const float*)d_in[43], (const float*)d_in[44],
                                 (const float*)d_in[49], (const float*)d_in[50],
                                 (float*)d_out + (size_t)t * 512,
                                 (float*)d_out + (size_t)64 * 512 + (size_t)t * 512);
  }
}
